// MultiConvAttn_30769145708728
// MI455X (gfx1250) — compile-verified, matrix operand order checked by reading, not run on hardware
//
#include <hip/hip_runtime.h>

typedef __attribute__((ext_vector_type(16))) _Float16 v16h;
typedef __attribute__((ext_vector_type(8)))  _Float16 v8h;
typedef __attribute__((ext_vector_type(8)))  float    v8f;
typedef __attribute__((ext_vector_type(4)))  float    v4f;

#define DM   1024
#define SEQ  1024
#define NB   32
#define NH   16
#define KW   3
#define NF   48          // NH*KW
#define LDS_STRIDE 1032  // 1024+8 f16: row stride 2064B = 516 dwords; 516%64==4 -> conflict-free, 16B aligned

static __device__ __forceinline__ v8f wmma_f16(v16h a, v16h b, v8f c) {
  return __builtin_amdgcn_wmma_f32_16x16x32_f16(false, a, false, b, (short)0, c, false, false);
}

static __device__ __forceinline__ v16h cvt16_f32(const float* __restrict__ p) {
  v16h r;
  const v4f* vp = (const v4f*)p;  // p is 64B aligned by construction
#pragma unroll
  for (int j = 0; j < 4; ++j) {
    v4f t = vp[j];
    r[4 * j + 0] = (_Float16)t.x;
    r[4 * j + 1] = (_Float16)t.y;
    r[4 * j + 2] = (_Float16)t.z;
    r[4 * j + 3] = (_Float16)t.w;
  }
  return r;
}

// ---------------------------------------------------------------------------
// Phase 1: qk = Q @ Wq + bq  (M=B*S=32768, N=48, K=1024), scatter-stored as
// f16 filters Wbuf[b][k][h][d]  (h = s>>6, 3d+k = (s&63)*48 + f).
// ---------------------------------------------------------------------------
__global__ __launch_bounds__(128) void mca_qk_kernel(
    const float* __restrict__ Q,   // (B*S, DM) f32
    const float* __restrict__ Wq,  // (DM, 48) f32
    const float* __restrict__ bq,  // (48) f32
    _Float16* __restrict__ Wbuf)   // (B, 3, 16, 1024) f16
{
  extern __shared__ _Float16 wql[];  // [48][LDS_STRIDE] f16, transposed Wq

  // cooperative transpose fill: Wq[d][f] (f32) -> wql[f][d] (f16)
  for (int idx = threadIdx.x; idx < DM * NF; idx += 128) {
    int d = idx / NF;
    int f = idx - d * NF;
    wql[f * LDS_STRIDE + d] = (_Float16)Wq[idx];
  }
  __syncthreads();

  const int lane = threadIdx.x & 31;
  const int wave = threadIdx.x >> 5;
  const int nl   = lane & 15;        // row (A) / col (B) within tile
  const int half = lane >> 4;        // K-chunk half select
  const int koff = half * 16;

  const int row0 = (blockIdx.x * 4 + wave) * 16;          // 16-row M tile
  const float* qrow = Q + (size_t)(row0 + nl) * DM;

  v8f c0 = {}, c1 = {}, c2 = {};

  for (int kc = 0; kc < DM; kc += 32) {
    v16h a = cvt16_f32(qrow + kc + koff);                 // A: 16 contig f32 -> f16

    const _Float16* bp0 = &wql[(0  + nl) * LDS_STRIDE + kc + koff];
    const _Float16* bp1 = &wql[(16 + nl) * LDS_STRIDE + kc + koff];
    const _Float16* bp2 = &wql[(32 + nl) * LDS_STRIDE + kc + koff];
    v16h fb0, fb1, fb2;
    {
      v8h l0 = *(const v8h*)bp0, h0 = *(const v8h*)(bp0 + 8);
      v8h l1 = *(const v8h*)bp1, h1 = *(const v8h*)(bp1 + 8);
      v8h l2 = *(const v8h*)bp2, h2 = *(const v8h*)(bp2 + 8);
#pragma unroll
      for (int i = 0; i < 8; ++i) {
        fb0[i] = l0[i]; fb0[i + 8] = h0[i];
        fb1[i] = l1[i]; fb1[i + 8] = h1[i];
        fb2[i] = l2[i]; fb2[i + 8] = h2[i];
      }
    }
    c0 = wmma_f16(a, fb0, c0);
    c1 = wmma_f16(a, fb1, c1);
    c2 = wmma_f16(a, fb2, c2);
  }

  // bias + scatter store of D tile.  D element (m,n): lane = (m<8?n:16+n), vgpr=m&7
  const float bias0 = bq[nl];
  const float bias1 = bq[16 + nl];
  const float bias2 = bq[32 + nl];

#pragma unroll
  for (int v = 0; v < 8; ++v) {
    int r  = row0 + half * 8 + v;    // global row = b*1024 + s
    int bb = r >> 10;
    int s  = r & 1023;
    int h  = s >> 6;
    int vbase = (s & 63) * NF;       // 3d+k = vbase + f

    float vals[3] = {c0[v] + bias0, c1[v] + bias1, c2[v] + bias2};
    int   fs[3]   = {nl, 16 + nl, 32 + nl};
#pragma unroll
    for (int t = 0; t < 3; ++t) {
      int vv = vbase + fs[t];
      int d  = vv / 3;
      int k  = vv - 3 * d;
      Wbuf[(((size_t)bb * KW + k) * NH + h) * DM + d] = (_Float16)vals[t];
    }
  }
}

// ---------------------------------------------------------------------------
// Phase 2: out[b,h,s] = sum_{k,d} Wbuf[b][k][h][d] * K[b, s+k-1, d]
// Per b: GEMM M=16(h), N=1024(s), K=3*1024.
// One 16-s tile per wave (2048 waves total) with two independent accumulator
// chains over even/odd 32-wide d-chunks for XDL pipelining.
// ---------------------------------------------------------------------------
__global__ __launch_bounds__(128) void mca_conv_kernel(
    const float* __restrict__ Kseq,      // (B, S, DM) f32
    const _Float16* __restrict__ Wbuf,   // (B, 3, 16, 1024) f16
    float* __restrict__ Out)             // (B, 16, S) f32
{
  const int b    = blockIdx.x;
  const int sblk = blockIdx.y;           // 0..15, 64 s per block
  const int lane = threadIdx.x & 31;
  const int wave = threadIdx.x >> 5;
  const int nl   = lane & 15;
  const int half = lane >> 4;
  const int koff = half * 16;

  const int s0 = sblk * 64 + wave * 16;  // this wave's 16-s N tile

  v8f cA = {}, cB = {};

  for (int k = 0; k < KW; ++k) {
    const _Float16* arow = Wbuf + (((size_t)b * KW + k) * NH + nl) * DM;  // A row h=nl
    const int t = s0 + nl + k - 1;       // key row for col nl
    const bool in = (t >= 0) && (t < SEQ);
    const float* krow = Kseq + ((size_t)b * SEQ + t) * DM;

    for (int dc = 0; dc < DM; dc += 64) {
      v16h a0 = *(const v16h*)(arow + dc + koff);        // 32B contig f16, 32B aligned
      v16h a1 = *(const v16h*)(arow + dc + 32 + koff);
      v16h fb0 = {}, fb1 = {};
      if (in) {
        fb0 = cvt16_f32(krow + dc + koff);
        fb1 = cvt16_f32(krow + dc + 32 + koff);
      }
      cA = wmma_f16(a0, fb0, cA);        // chain A: even chunks
      cB = wmma_f16(a1, fb1, cB);        // chain B: odd chunks
    }
  }

  // D tile: element (m=h, n=s-offset): lane = (m<8?n:16+n), vgpr = m&7
#pragma unroll
  for (int v = 0; v < 8; ++v) {
    int h = half * 8 + v;
    Out[(((size_t)b * NH + h) << 10) + s0 + nl] = cA[v] + cB[v];
  }
}

extern "C" void kernel_launch(void* const* d_in, const int* in_sizes, int n_in,
                              void* d_out, int out_size, void* d_ws, size_t ws_size,
                              hipStream_t stream) {
  (void)in_sizes; (void)n_in; (void)out_size;

  const float* Q   = (const float*)d_in[0];  // query_sequences (32,1024,1024)
  const float* Ks  = (const float*)d_in[1];  // key_sequences   (32,1024,1024)
  // d_in[2] value_sequences, d_in[3] key_value_sequence_lengths: unused by reference
  const float* Wq  = (const float*)d_in[4];  // (1024, 48)
  const float* bq  = (const float*)d_in[5];  // (48)

  _Float16* Wbuf = (_Float16*)d_ws;          // (32,3,16,1024) f16 = 3 MB
  if (ws_size < (size_t)NB * KW * NH * DM * sizeof(_Float16)) return;

  dim3 blk(128);
  size_t ldsBytes = (size_t)NF * LDS_STRIDE * sizeof(_Float16);  // ~97 KB of the 320 KB WGP LDS

  dim3 g1((NB * SEQ) / 64);                  // 512 workgroups, 4 waves x 16 rows each
  mca_qk_kernel<<<g1, blk, ldsBytes, stream>>>(Q, Wq, bq, Wbuf);

  dim3 g2(NB, SEQ / 64);                     // 32 x 16 workgroups, 2048 waves
  mca_conv_kernel<<<g2, blk, 0, stream>>>(Ks, Wbuf, (float*)d_out);
}
